// Transformer_28750511079499
// MI455X (gfx1250) — matrix operand order repaired and checked by reading, not compiled, not run on hardware
//
#include <hip/hip_runtime.h>
#include <math.h>

// ---------------- problem constants ----------------
#define D_    512
#define S_    512
#define B_    4
#define H_    8
#define DK_   64
#define DFF_  2048
#define V_    32000
#define NL_   6
#define ROWS_ (B_ * S_)          // 2048 tokens per stream (enc or dec)

typedef unsigned short u16;
typedef __attribute__((ext_vector_type(16))) __bf16        v16bf;
typedef __attribute__((ext_vector_type(8)))  float         v8f;
typedef __attribute__((ext_vector_type(8)))  unsigned int  v8u;

// ---------------- helpers ----------------
__device__ __forceinline__ u16 f2bf(float f) {
  unsigned int u = __builtin_bit_cast(unsigned int, f);
  u += 0x7fffu + ((u >> 16) & 1u);      // round-to-nearest-even
  return (u16)(u >> 16);
}

// ---------------- f32 [K,N] -> bf16 [N,K] tiled transpose-convert ----------------
// grid: (N/32, K/32, numMatrices); block: (32,8). K,N multiples of 32 (true for all weights).
__global__ void k_cvt_t(const float* __restrict__ in, u16* __restrict__ out, int K, int N) {
  __shared__ u16 tile[32][33];
  long mat = blockIdx.z;
  const float* ip = in  + mat * (long)K * N;
  u16*         op = out + mat * (long)K * N;
  int tx = threadIdx.x, ty = threadIdx.y;
  int n0 = blockIdx.x * 32, k0 = blockIdx.y * 32;
#pragma unroll
  for (int r = ty; r < 32; r += 8)
    tile[r][tx] = f2bf(ip[(long)(k0 + r) * N + n0 + tx]);   // coalesced read
  __syncthreads();
#pragma unroll
  for (int r = ty; r < 32; r += 8)
    op[(long)(n0 + r) * K + k0 + tx] = tile[tx][r];         // coalesced write
}

// ---------------- embedding + positional encoding ----------------
__global__ void k_embed(const int* __restrict__ tok, const float* __restrict__ emb,
                        const float* __restrict__ pe, float* __restrict__ out) {
  long i = (long)blockIdx.x * blockDim.x + threadIdx.x;  // [0, B*S*D)
  int d  = (int)(i % D_);
  long bs = i / D_;
  int s  = (int)(bs % S_);
  int t  = tok[bs];
  out[i] = emb[(long)t * D_ + d] * 22.62741699796952f /* sqrt(512) */ + pe[s * D_ + d];
}

// ---------------- layernorm (Bessel ddof=1, eps on std), writes bf16 ----------------
__global__ void k_layernorm(const float* __restrict__ x, const float* __restrict__ a,
                            const float* __restrict__ b, u16* __restrict__ out) {
  __shared__ float s1[256], s2[256];
  int row = blockIdx.x, t = threadIdx.x;          // 256 threads, 2 elems each
  const float* xr = x + (long)row * D_;
  float v0 = xr[t], v1 = xr[t + 256];
  s1[t] = v0 + v1;
  s2[t] = v0 * v0 + v1 * v1;
  __syncthreads();
  for (int o = 128; o > 0; o >>= 1) {
    if (t < o) { s1[t] += s1[t + o]; s2[t] += s2[t + o]; }
    __syncthreads();
  }
  float mean = s1[0] * (1.0f / D_);
  float var  = (s2[0] - (float)D_ * mean * mean) * (1.0f / (D_ - 1));
  float inv  = 1.0f / (sqrtf(var) + 1e-6f);
  u16* orow = out + (long)row * D_;
  orow[t]       = f2bf(a[t]       * (v0 - mean) * inv + b[t]);
  orow[t + 256] = f2bf(a[t + 256] * (v1 - mean) * inv + b[t + 256]);
}

// ---------------- softmax over width 512 (scores already scaled), writes bf16 ----------------
__global__ void k_softmax512(const float* __restrict__ sc, u16* __restrict__ attn) {
  __shared__ float sm[256];
  long row = blockIdx.x;                           // B*H*S rows
  const float* xr = sc + row * 512;
  int t = threadIdx.x;
  float a = xr[t], b = xr[t + 256];
  sm[t] = fmaxf(a, b);
  __syncthreads();
  for (int o = 128; o > 0; o >>= 1) { if (t < o) sm[t] = fmaxf(sm[t], sm[t + o]); __syncthreads(); }
  float m = sm[0];
  __syncthreads();
  float ea = __expf(a - m), eb = __expf(b - m);
  sm[t] = ea + eb;
  __syncthreads();
  for (int o = 128; o > 0; o >>= 1) { if (t < o) sm[t] += sm[t + o]; __syncthreads(); }
  float inv = 1.0f / sm[0];
  u16* orow = attn + row * 512;
  orow[t]       = f2bf(ea * inv);
  orow[t + 256] = f2bf(eb * inv);
}

// ---------------- log_softmax in place over width W (=32000) ----------------
__global__ void k_logsoftmax(float* __restrict__ y, int W) {
  __shared__ float sm[256];
  long row = blockIdx.x;
  float* xr = y + row * (long)W;
  int t = threadIdx.x;
  float m = -3.4e38f;
  for (int i = t; i < W; i += 256) m = fmaxf(m, xr[i]);
  sm[t] = m; __syncthreads();
  for (int o = 128; o > 0; o >>= 1) { if (t < o) sm[t] = fmaxf(sm[t], sm[t + o]); __syncthreads(); }
  m = sm[0];
  __syncthreads();
  float s = 0.f;
  for (int i = t; i < W; i += 256) s += __expf(xr[i] - m);
  sm[t] = s; __syncthreads();
  for (int o = 128; o > 0; o >>= 1) { if (t < o) sm[t] += sm[t + o]; __syncthreads(); }
  float ls = m + __logf(sm[0]);
  for (int i = t; i < W; i += 256) xr[i] -= ls;
}

// ---------------- bf16 WMMA GEMM:  C = scale*(A @ Bt^T) + bias + res ----------------
// A  : bf16 [M,K] row-major (lda), Bt : bf16 [N,K] row-major (ldb) -> both contiguous b128 frags.
// One wave computes a 32x64 output supertile: 2 A-frags x 4 B-frags -> 8 WMMAs per 32-K step.
// z = blockIdx.y split (zb,zh) = (z/Hdiv, z%Hdiv); operand offset = zb*SB + zh*SH
// (folds batch/head strides, incl. head column offsets).
// OT: store bf16 transposed (C[col*ldc + row]) - used to materialize V^T.
// __launch_bounds__(128,1): 4 waves/block, min-occupancy 1 -> large VGPR budget,
// keeps 8 f32 accumulators + 6 fragments in registers (no scratch spills).
template <bool OBF16, bool RELU, bool OT>
__global__ __launch_bounds__(128, 1)
void k_gemm(const u16* __restrict__ A, int lda, long aSB, long aSH,
            const u16* __restrict__ Bt, int ldb, long bSB, long bSH,
            void* C, int ldc, long cSB, long cSH,
            const float* __restrict__ bias,
            const float* res, int ldres,
            int M, int Nt /* N/64 */, int K, int Hdiv, float scale) {
  int lane = threadIdx.x & 31;
  int wave = threadIdx.x >> 5;
  int tile = blockIdx.x * (blockDim.x >> 5) + wave;   // wave-uniform
  int tM = tile / Nt;
  int tN = tile - tM * Nt;
  if (tM * 32 >= M) return;                           // uniform: EXEC all-ones at WMMA

  int z  = blockIdx.y;
  int zb = z / Hdiv, zh = z - zb * Hdiv;
  const u16* Ap = A  + zb * aSB + zh * aSH;
  const u16* Bp = Bt + zb * bSB + zh * bSH;
  long cOff = zb * cSB + zh * cSH;

  int ml = lane & 15;      // fragment row (A) / row of Bt (= output col)
  int kh = lane >> 4;      // K-half select

  const u16* aRow0 = Ap + (long)(tM * 32 + ml) * lda;
  const u16* aRow1 = aRow0 + 16 * (long)lda;
  const u16* bRow0 = Bp + (long)(tN * 64 + ml) * ldb;
  const u16* bRow1 = bRow0 + 16 * (long)ldb;
  const u16* bRow2 = bRow0 + 32 * (long)ldb;
  const u16* bRow3 = bRow0 + 48 * (long)ldb;

  v8f zacc = {};
  v8f acc[8];
#pragma unroll
  for (int t = 0; t < 8; t++) acc[t] = zacc;

#pragma unroll 2
  for (int kk = 0; kk < K; kk += 32) {
    int ka = kk + kh * 16;       // A fragment base (u16 units)
    int kb = kk + kh * 16;       // Bt fragment base
    const unsigned int* a0 = (const unsigned int*)(aRow0 + ka);
    const unsigned int* a1 = (const unsigned int*)(aRow1 + ka);
    const unsigned int* b0 = (const unsigned int*)(bRow0 + kb);
    const unsigned int* b1 = (const unsigned int*)(bRow1 + kb);
    const unsigned int* b2 = (const unsigned int*)(bRow2 + kb);
    const unsigned int* b3 = (const unsigned int*)(bRow3 + kb);
    // Prefetch next chunk using the SAME base pointers (+128B immediate offset):
    // folds into global_prefetch_b8 imm-offset form, no extra address math.
    // Speculative (SE-scope) prefetch past the end is silently dropped (ISA 10.5).
    __builtin_prefetch((const char*)a0 + 128, 0, 3);
    __builtin_prefetch((const char*)a1 + 128, 0, 3);
    __builtin_prefetch((const char*)b0 + 128, 0, 3);
    __builtin_prefetch((const char*)b1 + 128, 0, 3);
    __builtin_prefetch((const char*)b2 + 128, 0, 3);
    __builtin_prefetch((const char*)b3 + 128, 0, 3);
    v8u ar0, ar1, br0, br1, br2, br3;
    ar0[0] = a0[0]; ar0[1] = a0[1]; ar0[2] = a0[2]; ar0[3] = a0[3];
    ar0[4] = a0[4]; ar0[5] = a0[5]; ar0[6] = a0[6]; ar0[7] = a0[7];
    ar1[0] = a1[0]; ar1[1] = a1[1]; ar1[2] = a1[2]; ar1[3] = a1[3];
    ar1[4] = a1[4]; ar1[5] = a1[5]; ar1[6] = a1[6]; ar1[7] = a1[7];
#pragma unroll
    for (int t = 0; t < 8; t++) { br0[t] = b0[t]; br1[t] = b1[t]; br2[t] = b2[t]; br3[t] = b3[t]; }
    v16bf a0v = __builtin_bit_cast(v16bf, ar0);
    v16bf a1v = __builtin_bit_cast(v16bf, ar1);
    v16bf b0v = __builtin_bit_cast(v16bf, br0);
    v16bf b1v = __builtin_bit_cast(v16bf, br1);
    v16bf b2v = __builtin_bit_cast(v16bf, br2);
    v16bf b3v = __builtin_bit_cast(v16bf, br3);
    acc[0] = __builtin_amdgcn_wmma_f32_16x16x32_bf16(false, a0v, false, b0v, (short)0, acc[0], false, false);
    acc[1] = __builtin_amdgcn_wmma_f32_16x16x32_bf16(false, a0v, false, b1v, (short)0, acc[1], false, false);
    acc[2] = __builtin_amdgcn_wmma_f32_16x16x32_bf16(false, a0v, false, b2v, (short)0, acc[2], false, false);
    acc[3] = __builtin_amdgcn_wmma_f32_16x16x32_bf16(false, a0v, false, b3v, (short)0, acc[3], false, false);
    acc[4] = __builtin_amdgcn_wmma_f32_16x16x32_bf16(false, a1v, false, b0v, (short)0, acc[4], false, false);
    acc[5] = __builtin_amdgcn_wmma_f32_16x16x32_bf16(false, a1v, false, b1v, (short)0, acc[5], false, false);
    acc[6] = __builtin_amdgcn_wmma_f32_16x16x32_bf16(false, a1v, false, b2v, (short)0, acc[6], false, false);
    acc[7] = __builtin_amdgcn_wmma_f32_16x16x32_bf16(false, a1v, false, b3v, (short)0, acc[7], false, false);
  }

#pragma unroll
  for (int i = 0; i < 2; i++) {
#pragma unroll
    for (int j = 0; j < 4; j++) {
      v8f a = acc[i * 4 + j];
      int col = tN * 64 + j * 16 + ml;
      float bv = bias ? bias[col] : 0.0f;
#pragma unroll
      for (int r = 0; r < 8; r++) {
        int row = tM * 32 + i * 16 + kh * 8 + r;
        float v = a[r] * scale + bv;
        if (res)  v += res[(long)row * ldres + col];
        if (RELU) v  = fmaxf(v, 0.0f);
        if (OT)          ((u16*)C)[cOff + (long)col * ldc + row] = f2bf(v);
        else if (OBF16)  ((u16*)C)[cOff + (long)row * ldc + col] = f2bf(v);
        else             ((float*)C)[cOff + (long)row * ldc + col] = v;
      }
    }
  }
}

// ---------------- host-side GEMM dispatch ----------------
static inline void gemm_launch(hipStream_t st,
    const u16* A, int lda, long aSB, long aSH,
    const u16* Bt, int ldb, long bSB, long bSH,
    void* C, int ldc, long cSB, long cSH, bool obf16, bool ot,
    const float* bias, const float* res, int ldres, bool relu,
    int M, int N, int K, int Z, int Hdiv, float scale) {
  int Nt    = N / 64;
  int tiles = (M / 32) * Nt;
  dim3 grid((tiles + 3) / 4, Z, 1);
  dim3 blk(128, 1, 1);
  if (ot)
    k_gemm<true,  false, true ><<<grid, blk, 0, st>>>(A, lda, aSB, aSH, Bt, ldb, bSB, bSH,
        C, ldc, cSB, cSH, bias, res, ldres, M, Nt, K, Hdiv, scale);
  else if (obf16 && relu)
    k_gemm<true,  true,  false><<<grid, blk, 0, st>>>(A, lda, aSB, aSH, Bt, ldb, bSB, bSH,
        C, ldc, cSB, cSH, bias, res, ldres, M, Nt, K, Hdiv, scale);
  else if (obf16)
    k_gemm<true,  false, false><<<grid, blk, 0, st>>>(A, lda, aSB, aSH, Bt, ldb, bSB, bSH,
        C, ldc, cSB, cSH, bias, res, ldres, M, Nt, K, Hdiv, scale);
  else
    k_gemm<false, false, false><<<grid, blk, 0, st>>>(A, lda, aSB, aSH, Bt, ldb, bSB, bSH,
        C, ldc, cSB, cSH, bias, res, ldres, M, Nt, K, Hdiv, scale);
}

// ---------------- orchestration ----------------
extern "C" void kernel_launch(void* const* d_in, const int* in_sizes, int n_in,
                              void* d_out, int out_size, void* d_ws, size_t ws_size,
                              hipStream_t stream) {
  (void)in_sizes; (void)n_in; (void)out_size; (void)ws_size;

  const int*   src       = (const int*)  d_in[0];
  const int*   tgt       = (const int*)  d_in[1];
  const float* src_emb   = (const float*)d_in[2];
  const float* tgt_emb   = (const float*)d_in[3];
  const float* enc_qkv_w = (const float*)d_in[4];
  const float* enc_qkv_b = (const float*)d_in[5];
  const float* enc_out_w = (const float*)d_in[6];
  const float* enc_out_b = (const float*)d_in[7];
  const float* enc_ff1_w = (const float*)d_in[8];
  const float* enc_ff1_b = (const float*)d_in[9];
  const float* enc_ff2_w = (const float*)d_in[10];
  const float* enc_ff2_b = (const float*)d_in[11];
  const float* enc_ln_a  = (const float*)d_in[12];
  const float* enc_ln_bb = (const float*)d_in[13];
  const float* enc_fn_a  = (const float*)d_in[14];
  const float* enc_fn_b  = (const float*)d_in[15];
  const float* dec_sqkv_w= (const float*)d_in[16];
  const float* dec_sqkv_b= (const float*)d_in[17];
  const float* dec_sout_w= (const float*)d_in[18];
  const float* dec_sout_b= (const float*)d_in[19];
  const float* dec_cqkv_w= (const float*)d_in[20];
  const float* dec_cqkv_b= (const float*)d_in[21];
  const float* dec_cout_w= (const float*)d_in[22];
  const float* dec_cout_b= (const float*)d_in[23];
  const float* dec_ff1_w = (const float*)d_in[24];
  const float* dec_ff1_b = (const float*)d_in[25];
  const float* dec_ff2_w = (const float*)d_in[26];
  const float* dec_ff2_b = (const float*)d_in[27];
  const float* dec_ln_a  = (const float*)d_in[28];
  const float* dec_ln_bb = (const float*)d_in[29];
  const float* dec_fn_a  = (const float*)d_in[30];
  const float* dec_fn_b  = (const float*)d_in[31];
  const float* proj_w    = (const float*)d_in[32];
  const float* proj_b    = (const float*)d_in[33];
  const float* pe_src    = (const float*)d_in[34];
  const float* pe_tgt    = (const float*)d_in[35];

  // -------- workspace carve-out --------
  char*  wsp = (char*)d_ws;
  size_t off = 0;
  auto alloc = [&](size_t bytes) -> void* {
    void* p = wsp + off;
    off += (bytes + 255) & ~(size_t)255;
    return p;
  };
  const size_t DD   = (size_t)D_ * D_;
  const size_t DFFD = (size_t)D_ * DFF_;

  // bf16 TRANSPOSED weights (each slice stored [N,K] row-major)
  u16* wb_eqkv = (u16*)alloc(NL_ * 3 * DD * 2);
  u16* wb_eout = (u16*)alloc(NL_ * DD * 2);
  u16* wb_eff1 = (u16*)alloc(NL_ * DFFD * 2);
  u16* wb_eff2 = (u16*)alloc(NL_ * DFFD * 2);
  u16* wb_sqkv = (u16*)alloc(NL_ * 3 * DD * 2);
  u16* wb_sout = (u16*)alloc(NL_ * DD * 2);
  u16* wb_cqkv = (u16*)alloc(NL_ * 3 * DD * 2);
  u16* wb_cout = (u16*)alloc(NL_ * DD * 2);
  u16* wb_dff1 = (u16*)alloc(NL_ * DFFD * 2);
  u16* wb_dff2 = (u16*)alloc(NL_ * DFFD * 2);
  u16* wb_proj = (u16*)alloc((size_t)D_ * V_ * 2);     // [V, D]
  // activations
  float* xe   = (float*)alloc((size_t)ROWS_ * D_ * 4);     // encoder stream (f32)
  float* yd   = (float*)alloc((size_t)ROWS_ * D_ * 4);     // decoder stream (f32)
  u16*  encb  = (u16*)  alloc((size_t)ROWS_ * D_ * 2);     // final-encoder LN (bf16)
  u16*  hb    = (u16*)  alloc((size_t)ROWS_ * D_ * 2);     // LN output (bf16)
  u16*  qb    = (u16*)  alloc((size_t)ROWS_ * D_ * 2);
  u16*  kbuf  = (u16*)  alloc((size_t)ROWS_ * D_ * 2);
  u16*  vT    = (u16*)  alloc((size_t)ROWS_ * D_ * 2);     // V^T: [h*DK+d, token]
  u16*  aob   = (u16*)  alloc((size_t)ROWS_ * D_ * 2);     // attn output pre-proj
  u16*  ffb   = (u16*)  alloc((size_t)ROWS_ * DFF_ * 2);   // relu(ff1) (bf16)
  float* scor = (float*)alloc((size_t)B_ * H_ * S_ * S_ * 4);
  u16*  attnb = (u16*)  alloc((size_t)B_ * H_ * S_ * S_ * 2);

  // transpose-convert all weights (one pass, deterministic, graph-safe)
  auto cvtT = [&](const float* in, u16* out, int K, int N, int count) {
    k_cvt_t<<<dim3(N / 32, K / 32, count), dim3(32, 8), 0, stream>>>(in, out, K, N);
  };
  cvtT(enc_qkv_w, wb_eqkv, D_, D_, NL_ * 3);
  cvtT(enc_out_w, wb_eout, D_, D_, NL_);
  cvtT(enc_ff1_w, wb_eff1, D_, DFF_, NL_);
  cvtT(enc_ff2_w, wb_eff2, DFF_, D_, NL_);
  cvtT(dec_sqkv_w, wb_sqkv, D_, D_, NL_ * 3);
  cvtT(dec_sout_w, wb_sout, D_, D_, NL_);
  cvtT(dec_cqkv_w, wb_cqkv, D_, D_, NL_ * 3);
  cvtT(dec_cout_w, wb_cout, D_, D_, NL_);
  cvtT(dec_ff1_w, wb_dff1, D_, DFF_, NL_);
  cvtT(dec_ff2_w, wb_dff2, DFF_, D_, NL_);
  cvtT(proj_w,    wb_proj, D_, V_, 1);

  auto ln = [&](const float* x, const float* a, const float* b, u16* out) {
    k_layernorm<<<dim3(ROWS_), dim3(256), 0, stream>>>(x, a, b, out);
  };

  const long SD  = (long)S_ * D_;
  const long SS  = (long)S_ * S_;
  const long HSS = (long)H_ * SS;

  // multi-head attention: xq/xkv bf16 [2048,512]; residual-accumulates into xres (f32)
  auto mha = [&](const u16* xq, const u16* xkv,
                 const u16* WqkvT, const float* Bqkv,
                 const u16* WoutT, const float* Bout, float* xres) {
    // Q, K: bf16 row-major; V: stored transposed (OT) as [h*DK+d, token]
    gemm_launch(stream, xq,  D_, 0, 0, WqkvT + 0 * DD, D_, 0, 0,
                qb, D_, 0, 0, true, false, Bqkv + 0 * D_, nullptr, 0, false,
                ROWS_, D_, D_, 1, 1, 1.0f);
    gemm_launch(stream, xkv, D_, 0, 0, WqkvT + 1 * DD, D_, 0, 0,
                kbuf, D_, 0, 0, true, false, Bqkv + 1 * D_, nullptr, 0, false,
                ROWS_, D_, D_, 1, 1, 1.0f);
    gemm_launch(stream, xkv, D_, 0, 0, WqkvT + 2 * DD, D_, 0, 0,
                vT, ROWS_, 0, 0, true, true, Bqkv + 2 * D_, nullptr, 0, false,
                ROWS_, D_, D_, 1, 1, 1.0f);
    // scores[b,h,q,k] = (Q . K) / sqrt(DK)   (Bt = K activations, contiguous)
    gemm_launch(stream, qb, D_, SD, DK_, kbuf, D_, SD, DK_,
                scor, S_, HSS, SS, false, false, nullptr, nullptr, 0, false,
                S_, S_, DK_, B_ * H_, H_, 0.125f);
    k_softmax512<<<dim3(B_ * H_ * S_), dim3(256), 0, stream>>>(scor, attnb);
    // O[b,q,h,d] = attn @ V    (Bt = V^T slice: row=d, contiguous over keys)
    gemm_launch(stream, attnb, S_, HSS, SS, vT, ROWS_, (long)S_, (long)DK_ * ROWS_,
                aob, D_, SD, DK_, true, false, nullptr, nullptr, 0, false,
                S_, DK_, S_, B_ * H_, H_, 1.0f);
    // out-projection + bias + residual -> xres (f32)
    gemm_launch(stream, aob, D_, 0, 0, WoutT, D_, 0, 0,
                xres, D_, 0, 0, false, false, Bout, xres, D_, false,
                ROWS_, D_, D_, 1, 1, 1.0f);
  };

  auto ffn = [&](const u16* h, const u16* W1T, const float* B1,
                 const u16* W2T, const float* B2, float* xres) {
    gemm_launch(stream, h, D_, 0, 0, W1T, D_, 0, 0,
                ffb, DFF_, 0, 0, true, false, B1, nullptr, 0, true,   // fused ReLU
                ROWS_, DFF_, D_, 1, 1, 1.0f);
    gemm_launch(stream, ffb, DFF_, 0, 0, W2T, DFF_, 0, 0,
                xres, D_, 0, 0, false, false, B2, xres, D_, false,
                ROWS_, D_, DFF_, 1, 1, 1.0f);
  };

  // -------- encoder --------
  k_embed<<<dim3(4096), dim3(256), 0, stream>>>(src, src_emb, pe_src, xe);
  for (int i = 0; i < NL_; i++) {
    ln(xe, enc_ln_a + (size_t)(i * 2 + 0) * D_, enc_ln_bb + (size_t)(i * 2 + 0) * D_, hb);
    mha(hb, hb, wb_eqkv + (size_t)i * 3 * DD, enc_qkv_b + (size_t)i * 3 * D_,
        wb_eout + (size_t)i * DD, enc_out_b + (size_t)i * D_, xe);
    ln(xe, enc_ln_a + (size_t)(i * 2 + 1) * D_, enc_ln_bb + (size_t)(i * 2 + 1) * D_, hb);
    ffn(hb, wb_eff1 + (size_t)i * DFFD, enc_ff1_b + (size_t)i * DFF_,
        wb_eff2 + (size_t)i * DFFD, enc_ff2_b + (size_t)i * D_, xe);
  }
  ln(xe, enc_fn_a, enc_fn_b, encb);

  // -------- decoder --------
  k_embed<<<dim3(4096), dim3(256), 0, stream>>>(tgt, tgt_emb, pe_tgt, yd);
  for (int i = 0; i < NL_; i++) {
    ln(yd, dec_ln_a + (size_t)(i * 3 + 0) * D_, dec_ln_bb + (size_t)(i * 3 + 0) * D_, hb);
    mha(hb, hb, wb_sqkv + (size_t)i * 3 * DD, dec_sqkv_b + (size_t)i * 3 * D_,
        wb_sout + (size_t)i * DD, dec_sout_b + (size_t)i * D_, yd);
    ln(yd, dec_ln_a + (size_t)(i * 3 + 1) * D_, dec_ln_bb + (size_t)(i * 3 + 1) * D_, hb);
    mha(hb, encb, wb_cqkv + (size_t)i * 3 * DD, dec_cqkv_b + (size_t)i * 3 * D_,
        wb_cout + (size_t)i * DD, dec_cout_b + (size_t)i * D_, yd);
    ln(yd, dec_ln_a + (size_t)(i * 3 + 2) * D_, dec_ln_bb + (size_t)(i * 3 + 2) * D_, hb);
    ffn(hb, wb_dff1 + (size_t)i * DFFD, dec_ff1_b + (size_t)i * DFF_,
        wb_dff2 + (size_t)i * DFFD, dec_ff2_b + (size_t)i * D_, yd);
  }
  ln(yd, dec_fn_a, dec_fn_b, hb);   // hb = final decoder LN (bf16)

  // -------- vocab projection + log_softmax (into d_out, f32) --------
  gemm_launch(stream, hb, D_, 0, 0, wb_proj, D_, 0, 0,
              d_out, V_, 0, 0, false, false, proj_b, nullptr, 0, false,
              ROWS_, V_, D_, 1, 1, 1.0f);
  k_logsoftmax<<<dim3(ROWS_), dim3(256), 0, stream>>>((float*)d_out, V_);
}
